// DGCNN_15264313770133
// MI455X (gfx1250) — matrix operand order repaired and checked by reading, not compiled, not run on hardware
//
#include <hip/hip_runtime.h>

// ---------------------------------------------------------------------------
// DGCNN forward for MI455X (gfx1250, wave32, WMMA).
//   Phase 1 (prep):   W1..W4 f32 -> f16 row-major (W4 padded to 48x64), b4 pad.
//   Phase 2 (mlp):    fused 4-layer MLP on ONLY the 30 (+2 pad) needed nodes
//                     per graph, computed transposed (Y = H^T) so WMMA D
//                     fragments re-pack into next-layer B fragments with
//                     v_cvt_pkrtz + shfl_xor(16) -- no LDS between layers.
//                     Mean-pool over 30 nodes done in-register via shfl_xor.
//   Phase 3 (head):   conv1(k5)+relu+maxpool2 -> conv2(k5)+relu -> FC(2),
//                     one wave per graph, weights staged in LDS.
// ---------------------------------------------------------------------------

typedef _Float16 f16_t;
typedef __attribute__((ext_vector_type(16))) _Float16 v16h;
typedef __attribute__((ext_vector_type(8)))  _Float16 v8h;
typedef __attribute__((ext_vector_type(8)))  float    v8f;
typedef __attribute__((ext_vector_type(4)))  float    v4f;

#define NGRAPH 10000
#define NODES_PER_G 100
#define FD 64

// d_ws byte layout
#define WS_WH_OFF   0        // 15360 halfs: W1(64x64) W2(64x64) W3(64x64) W4pad(48x64)
#define WS_B4_OFF   30720    // 48 floats (b4 padded with zeros)
#define WS_POOL_OFF 32768    // NGRAPH*34 floats (pooled, sort-pool mean)

__device__ __forceinline__ unsigned pk2(float a, float b) {
    return __builtin_bit_cast(unsigned, __builtin_amdgcn_cvt_pkrtz(a, b));
}
__device__ __forceinline__ v8f wmma32f16(v16h a, v16h b, v8f c) {
    return __builtin_amdgcn_wmma_f32_16x16x32_f16(false, a, false, b, (short)0, c,
                                                  false, false);
}

// ---------------------------- phase 1: prep -------------------------------
__global__ __launch_bounds__(256) void dgcnn_prep(
    const float* __restrict__ W1, const float* __restrict__ W2,
    const float* __restrict__ W3, const float* __restrict__ W4,
    const float* __restrict__ b4,
    f16_t* __restrict__ wh, float* __restrict__ b4pad)
{
    const int i = blockIdx.x * 256 + threadIdx.x;
    if (i < 15360) {
        float v;
        if (i < 4096)       v = W1[i];
        else if (i < 8192)  v = W2[i - 4096];
        else if (i < 12288) v = W3[i - 8192];
        else {
            const int j = i - 12288;
            v = ((j >> 6) < 34) ? W4[j] : 0.0f;   // pad rows 34..47 with zeros
        }
        wh[i] = (f16_t)v;
    }
    const int k = i - 15360;
    if (k >= 0 && k < 48) b4pad[k] = (k < 34) ? b4[k] : 0.0f;
}

// --------------------- phase 2: fused MLP + sort-pool ---------------------
// one wave per graph; 8 waves / block; 60 v_wmma per wave
__global__ __launch_bounds__(256) void dgcnn_mlp_pool(
    const float* __restrict__ x,
    const float* __restrict__ b1, const float* __restrict__ b2,
    const float* __restrict__ b3,
    const f16_t* __restrict__ wh, const float* __restrict__ b4pad,
    float* __restrict__ pooled)
{
    const int lane = threadIdx.x & 31;
    const int wv   = threadIdx.x >> 5;
    const int g    = blockIdx.x * 8 + wv;
    const int l16  = lane & 15;   // node column within tile / weight row
    const int hf   = lane >> 4;   // half-wave select

    // --- load X^T tiles directly as WMMA B fragments (f32 -> packed f16) ---
    // B 32x16 f16 fragment in this code's contiguous k order (same as A): lanes 0-15 hold K=k0..k0+15 (2/VGPR), lanes 16-31 K+16.
    v16h Bin[2][2];
#pragma unroll
    for (int t = 0; t < 2; ++t) {
#pragma unroll
        for (int kt = 0; kt < 2; ++kt) {
            const float* xr = x + (size_t)(g * NODES_PER_G + t * 16 + l16) * FD
                                + kt * 32 + hf * 16;
            v4f q0 = *(const v4f*)(xr + 0);
            v4f q1 = *(const v4f*)(xr + 4);
            v4f q2 = *(const v4f*)(xr + 8);
            v4f q3 = *(const v4f*)(xr + 12);
            union { v16h v; unsigned u[8]; } bb;
            bb.u[0] = pk2(q0[0], q0[1]); bb.u[1] = pk2(q0[2], q0[3]);
            bb.u[2] = pk2(q1[0], q1[1]); bb.u[3] = pk2(q1[2], q1[3]);
            bb.u[4] = pk2(q2[0], q2[1]); bb.u[5] = pk2(q2[2], q2[3]);
            bb.u[6] = pk2(q3[0], q3[1]); bb.u[7] = pk2(q3[2], q3[3]);
            Bin[t][kt] = bb.v;
        }
    }

    // pooling mask: tile t=1 columns 14,15 are node pos 30,31 -> excluded
    const float tmask1 = (l16 >= 14) ? 0.0f : 1.0f;

    v8f psum[3];
#pragma unroll
    for (int mt = 0; mt < 3; ++mt)
#pragma unroll
        for (int r = 0; r < 8; ++r) psum[mt][r] = 0.0f;

#pragma unroll
    for (int l = 0; l < 4; ++l) {
        const f16_t* W    = wh + l * 4096;
        const float* bias = (l == 0) ? b1 : (l == 1) ? b2 : (l == 2) ? b3 : b4pad;
        unsigned P[2][4][4];   // packed f16 D tiles (t, mtile, pair)
#pragma unroll
        for (int mt = 0; mt < 4; ++mt) {
            if (l == 3 && mt == 3) continue;   // layer 4: only 34(->48) out chans
            const int m0 = mt * 16;
            // A 16x32 f16 fragment of W: lane row = m0+l16,
            // per-lane 8-half chunks at k0 + 16*hf and k0 + 16*hf + 8
            const f16_t* wr = W + (size_t)(m0 + l16) * FD;
            union { v16h v; v8h h[2]; } A0, A1;
            A0.h[0] = *(const v8h*)(wr + 0  + hf * 16);
            A0.h[1] = *(const v8h*)(wr + 8  + hf * 16);
            A1.h[0] = *(const v8h*)(wr + 32 + hf * 16);
            A1.h[1] = *(const v8h*)(wr + 40 + hf * 16);
            v8f bias8;
#pragma unroll
            for (int r = 0; r < 8; ++r) bias8[r] = bias[m0 + hf * 8 + r];
#pragma unroll
            for (int t = 0; t < 2; ++t) {
                v8f c = bias8;
                c = wmma32f16(A0.v, Bin[t][0], c);   // K = 0..31
                c = wmma32f16(A1.v, Bin[t][1], c);   // K = 32..63
#pragma unroll
                for (int r = 0; r < 8; ++r) c[r] = fmaxf(c[r], 0.0f);  // relu
                if (l == 3) {
                    const float m = (t == 1) ? tmask1 : 1.0f;
#pragma unroll
                    for (int r = 0; r < 8; ++r) psum[mt][r] += c[r] * m;
                } else {
#pragma unroll
                    for (int j = 0; j < 4; ++j)
                        P[t][mt][j] = pk2(c[2 * j], c[2 * j + 1]);
                }
            }
        }
        if (l < 3) {
            // Re-pack D tiles into next-layer B fragments:
            //  lane<16 : halves 0..7 = tile(2kt) self, 8..15 = tile(2kt) partner
            //  lane>=16: halves 0..7 = tile(2kt+1) partner, 8..15 = self
#pragma unroll
            for (int t = 0; t < 2; ++t) {
#pragma unroll
                for (int kt = 0; kt < 2; ++kt) {
                    union { v16h v; unsigned u[8]; } nb;
#pragma unroll
                    for (int j = 0; j < 4; ++j) {
                        const unsigned lo  = P[t][2 * kt][j];
                        const unsigned hi  = P[t][2 * kt + 1][j];
                        const unsigned lox = __shfl_xor(lo, 16, 32);
                        const unsigned hix = __shfl_xor(hi, 16, 32);
                        nb.u[j]     = hf ? hix : lo;
                        nb.u[4 + j] = hf ? hi  : lox;
                    }
                    Bin[t][kt] = nb.v;
                }
            }
        }
    }

    // --- mean pool over the 30 selected nodes (columns == lanes) ---
#pragma unroll
    for (int mt = 0; mt < 3; ++mt) {
#pragma unroll
        for (int r = 0; r < 8; ++r) {
            float s = psum[mt][r];
#pragma unroll
            for (int off = 1; off < 16; off <<= 1) s += __shfl_xor(s, off, 32);
            psum[mt][r] = s;
        }
        if (l16 == 0) {   // lane 0 -> chans m0..m0+7, lane 16 -> m0+8..m0+15
#pragma unroll
            for (int r = 0; r < 8; ++r) {
                const int ch = mt * 16 + hf * 8 + r;
                if (ch < 34)
                    pooled[(size_t)g * 34 + ch] = psum[mt][r] * (1.0f / 30.0f);
            }
        }
    }
}

// ------------------------- phase 3: conv head -----------------------------
__global__ __launch_bounds__(256) void dgcnn_conv_head(
    const float* __restrict__ pooled,
    const float* __restrict__ cw1, const float* __restrict__ cb1,
    const float* __restrict__ cw2, const float* __restrict__ cb2,
    const float* __restrict__ Wo,  const float* __restrict__ bo,
    float* __restrict__ out)
{
    __shared__ float s_w1[80], s_b1[16], s_w2[2560], s_b2[32], s_wo[704], s_bo[2];
    __shared__ float s_p[8][34];
    __shared__ float s_q[8][240];

    const int tid = threadIdx.x;
    for (int i = tid; i < 80;   i += 256) s_w1[i] = cw1[i];
    for (int i = tid; i < 16;   i += 256) s_b1[i] = cb1[i];
    for (int i = tid; i < 2560; i += 256) s_w2[i] = cw2[i];
    for (int i = tid; i < 32;   i += 256) s_b2[i] = cb2[i];
    for (int i = tid; i < 704;  i += 256) s_wo[i] = Wo[i];
    if (tid < 2) s_bo[tid] = bo[tid];

    const int lane = tid & 31;
    const int wv   = tid >> 5;
    const int g    = blockIdx.x * 8 + wv;
    for (int i = lane; i < 34; i += 32) s_p[wv][i] = pooled[(size_t)g * 34 + i];
    __syncthreads();

    // conv1 (16x1x5, VALID -> 30) + relu + maxpool2 -> [16][15]
    for (int idx = lane; idx < 240; idx += 32) {
        const int c = idx / 15, pp = idx % 15;
        float a = s_b1[c], b = s_b1[c];
#pragma unroll
        for (int j = 0; j < 5; ++j) {
            const float w = s_w1[c * 5 + j];
            a += s_p[wv][2 * pp + j]     * w;
            b += s_p[wv][2 * pp + 1 + j] * w;
        }
        s_q[wv][idx] = fmaxf(fmaxf(a, b), 0.0f);
    }
    __syncthreads();

    // conv2 (32x16x5, VALID -> 11) + relu, fused with final FC (2 x 352)
    float a0 = 0.0f, a1 = 0.0f;
    for (int idx = lane; idx < 352; idx += 32) {
        const int oc = idx / 11, pos = idx % 11;
        float v = s_b2[oc];
#pragma unroll
        for (int ic = 0; ic < 16; ++ic) {
#pragma unroll
            for (int j = 0; j < 5; ++j)
                v += s_q[wv][ic * 15 + pos + j] * s_w2[(oc * 16 + ic) * 5 + j];
        }
        v = fmaxf(v, 0.0f);
        a0 += v * s_wo[idx];
        a1 += v * s_wo[352 + idx];
    }
#pragma unroll
    for (int off = 1; off < 32; off <<= 1) {
        a0 += __shfl_xor(a0, off, 32);
        a1 += __shfl_xor(a1, off, 32);
    }
    if (lane == 0) {
        out[(size_t)g * 2 + 0] = a0 + s_bo[0];
        out[(size_t)g * 2 + 1] = a1 + s_bo[1];
    }
}

// ------------------------------ launcher ----------------------------------
extern "C" void kernel_launch(void* const* d_in, const int* in_sizes, int n_in,
                              void* d_out, int out_size, void* d_ws, size_t ws_size,
                              hipStream_t stream) {
    const float* x   = (const float*)d_in[0];
    // d_in[1] edge_index, d_in[2] batch: unused by the reference math
    const float* W1  = (const float*)d_in[3];
    const float* b1  = (const float*)d_in[4];
    const float* W2  = (const float*)d_in[5];
    const float* b2  = (const float*)d_in[6];
    const float* W3  = (const float*)d_in[7];
    const float* b3  = (const float*)d_in[8];
    const float* W4  = (const float*)d_in[9];
    const float* b4  = (const float*)d_in[10];
    const float* cw1 = (const float*)d_in[11];
    const float* cb1 = (const float*)d_in[12];
    const float* cw2 = (const float*)d_in[13];
    const float* cb2 = (const float*)d_in[14];
    const float* Wo  = (const float*)d_in[15];
    const float* bo  = (const float*)d_in[16];

    f16_t* wh     = (f16_t*)((char*)d_ws + WS_WH_OFF);
    float* b4pad  = (float*)((char*)d_ws + WS_B4_OFF);
    float* pooled = (float*)((char*)d_ws + WS_POOL_OFF);
    float* out    = (float*)d_out;

    dgcnn_prep<<<61, 256, 0, stream>>>(W1, W2, W3, W4, b4, wh, b4pad);
    dgcnn_mlp_pool<<<NGRAPH / 8, 256, 0, stream>>>(x, b1, b2, b3, wh, b4pad, pooled);
    dgcnn_conv_head<<<NGRAPH / 8, 256, 0, stream>>>(pooled, cw1, cb1, cw2, cb2,
                                                    Wo, bo, out);
}
